// MQA_83820581749268
// MI455X (gfx1250) — matrix operand order repaired and checked by reading, not compiled, not run on hardware
//
#include <hip/hip_runtime.h>

// ---------------------------------------------------------------------------
// CDNA5 (gfx1250) MQA forward, bf16 WMMA everywhere, flash-style attention.
// ---------------------------------------------------------------------------

typedef __attribute__((ext_vector_type(16))) __bf16        v16bf;
typedef __attribute__((ext_vector_type(8)))  float         v8f;
typedef __attribute__((ext_vector_type(4)))  unsigned int  v4u;

union BFrag { v4u q[2]; v16bf v; };   // 32 bytes = 16 bf16 = one WMMA A/B fragment per lane

__device__ __forceinline__ unsigned short f2bf(float f) {
  unsigned int u = __float_as_uint(f);
  u += 0x7fffu + ((u >> 16) & 1u);          // round-to-nearest-even
  return (unsigned short)(u >> 16);
}

__device__ __forceinline__ v8f wmma_bf16(v16bf a, v16bf b, v8f c) {
  // D = A(16x32 bf16) * B(32x16 bf16) + C(16x16 f32)
  return __builtin_amdgcn_wmma_f32_16x16x32_bf16(false, a, false, b, (short)0, c,
                                                 false, false);
}

__device__ __forceinline__ v8f v8f_zero() {
  v8f r;
#pragma unroll
  for (int g = 0; g < 8; ++g) r[g] = 0.0f;
  return r;
}

// ---------------------------------------------------------------------------
// Elementwise f32 -> bf16
// ---------------------------------------------------------------------------
__global__ void k_f32_to_bf16(const float* __restrict__ s,
                              unsigned short* __restrict__ d, long long n) {
  long long i = (long long)blockIdx.x * blockDim.x + threadIdx.x;
  if (i < n) d[i] = f2bf(s[i]);
}

// cache_k [b,1,CACHE,D] f32 -> Kb [b,KV,D] bf16 (rows 0..CACHE-1)
__global__ void k_cache_to_K(const float* __restrict__ s,
                             unsigned short* __restrict__ Kb,
                             int CACHE, int D, int KV, long long n) {
  long long i = (long long)blockIdx.x * blockDim.x + threadIdx.x;
  if (i >= n) return;
  int dc = (int)(i % D);
  long long r = i / D;
  int sp = (int)(r % CACHE);
  int bi = (int)(r / CACHE);
  Kb[((long long)bi * KV + sp) * D + dc] = f2bf(s[i]);
}

// cache_v [b,1,CACHE,D] f32 -> Vt [b,D,KV] bf16 (cols 0..CACHE-1), transposed
__global__ void k_cache_to_Vt(const float* __restrict__ s,
                              unsigned short* __restrict__ Vt,
                              int CACHE, int D, int KV, long long n) {
  long long i = (long long)blockIdx.x * blockDim.x + threadIdx.x;
  if (i >= n) return;
  int dc = (int)(i % D);
  long long r = i / D;
  int sp = (int)(r % CACHE);
  int bi = (int)(r / CACHE);
  Vt[((long long)bi * D + dc) * KV + sp] = f2bf(s[i]);
}

// ---------------------------------------------------------------------------
// Generic NT GEMM:  C[m,n] = sum_k A[m,k] * Bw[n,k]    (torch Linear y = x W^T)
// A, Bw are bf16 row-major.  Output indexed as
//   dst = rowBase + (r/rpb)*batchStride + (r%rpb)*rowStride + c*colStride
// Block = 256 threads (8 waves) -> 64(M) x 256(N); wave tile 32 x 64.
// ---------------------------------------------------------------------------
template <bool OUT_F32>
__global__ void __launch_bounds__(256, 1)
gemm_bf16_wmma(const unsigned short* __restrict__ A, int lda,
               const unsigned short* __restrict__ Bw, int ldb,
               void* __restrict__ Cout,
               int M, int N, int K,
               int rpb, long long batchStride, long long rowBase,
               long long rowStride, long long colStride) {
  const int tid  = threadIdx.x;
  const int wave = tid >> 5;
  const int lane = tid & 31;
  const int lh   = lane >> 4;     // 16-lane half
  const int lr   = lane & 15;
  const int mBase = blockIdx.x * 64  + (wave >> 2) * 32;
  const int nBase = blockIdx.y * 256 + (wave & 3) * 64;
  if (mBase >= M || nBase >= N) return;

  v8f acc[2][4];
#pragma unroll
  for (int m = 0; m < 2; ++m)
#pragma unroll
    for (int n = 0; n < 4; ++n) acc[m][n] = v8f_zero();

  for (int k0 = 0; k0 < K; k0 += 32) {
    if (k0 + 32 < K) {
      __builtin_prefetch(A  + (size_t)(mBase + lr) * lda + k0 + 32, 0, 1);
      __builtin_prefetch(Bw + (size_t)(nBase + lr) * ldb + k0 + 32, 0, 1);
    }
    // A fragments: lane holds row (mBase+m*16+lr), K chunks {lh*8..} and {16+lh*8..}
    BFrag a[2];
#pragma unroll
    for (int m = 0; m < 2; ++m) {
      const unsigned short* ap = A + (size_t)(mBase + m * 16 + lr) * lda + k0;
      a[m].q[0] = *(const v4u*)(ap + lh * 8);
      a[m].q[1] = *(const v4u*)(ap + 16 + lh * 8);
    }
    // B fragments: lane holds weight row (nBase+n*16+lr), K chunks {lh*8..} and {16+lh*8..}
    BFrag b[4];
#pragma unroll
    for (int n = 0; n < 4; ++n) {
      const unsigned short* bp = Bw + (size_t)(nBase + n * 16 + lr) * ldb + k0 + lh * 8;
      b[n].q[0] = *(const v4u*)(bp);
      b[n].q[1] = *(const v4u*)(bp + 16);
    }
#pragma unroll
    for (int m = 0; m < 2; ++m)
#pragma unroll
      for (int n = 0; n < 4; ++n)
        acc[m][n] = wmma_bf16(a[m].v, b[n].v, acc[m][n]);
  }

  // C/D layout: VGPR g, lane -> (row = g + 8*lh, col = lr)
#pragma unroll
  for (int m = 0; m < 2; ++m)
#pragma unroll
    for (int n = 0; n < 4; ++n)
#pragma unroll
      for (int g = 0; g < 8; ++g) {
        int r = mBase + m * 16 + g + 8 * lh;
        int c = nBase + n * 16 + lr;
        long long dst = rowBase + (long long)(r / rpb) * batchStride +
                        (long long)(r % rpb) * rowStride + (long long)c * colStride;
        if (OUT_F32) ((float*)Cout)[dst] = acc[m][n][g];
        else ((unsigned short*)Cout)[dst] = f2bf(acc[m][n][g]);
      }
}

// ---------------------------------------------------------------------------
// Fused MQA flash attention.
//   Q  [b*T, H*D] bf16 (head-major cols), Kb [b,KV,D] bf16, Vt [b,D,KV] bf16
//   Y  [b*T, H*D] bf16
// grid = (T/64, H, B), block = 128 (4 waves); each wave owns one 16-row q-tile.
// kv processed in blocks of 32: 8 score WMMAs + online softmax + 8 PV WMMAs.
// P (16x32) is re-shaped C-layout -> A-layout through a private 1KB LDS tile.
// __launch_bounds__(128,1): whole VGPR file available -> no scratch spills of
// the persistent Q fragments / accumulators in the kv loop.
// ---------------------------------------------------------------------------
__global__ void __launch_bounds__(128, 1)
mqa_flash_attn(const unsigned short* __restrict__ Q,
               const unsigned short* __restrict__ Kb,
               const unsigned short* __restrict__ Vt,
               unsigned short* __restrict__ Y,
               int T, int CACHE, int KV, int D, int HD) {
  __shared__ __align__(16) unsigned short Pbuf[4][16 * 32];

  const int wave = threadIdx.x >> 5;
  const int lane = threadIdx.x & 31;
  const int lh = lane >> 4;
  const int lr = lane & 15;
  const int bi = blockIdx.z;
  const int hi = blockIdx.y;
  const int qt = blockIdx.x * 4 + wave;       // 16-row query tile index
  const int qrow0 = qt * 16;

  const unsigned short* qbase = Q + ((size_t)bi * T + qrow0) * HD + (size_t)hi * D;
  const unsigned short* Kbase = Kb + (size_t)bi * KV * D;
  const unsigned short* Vbase = Vt + (size_t)bi * D * KV;
  unsigned short* P = &Pbuf[wave][0];

  // Q fragments for the 4 K-steps over d (D = 128 = 4 * 32)
  BFrag qa[4];
#pragma unroll
  for (int kk = 0; kk < 4; ++kk) {
    const unsigned short* qp = qbase + (size_t)lr * HD + kk * 32;
    qa[kk].q[0] = *(const v4u*)(qp + lh * 8);
    qa[kk].q[1] = *(const v4u*)(qp + 16 + lh * 8);
  }

  v8f acc[8];                                  // 16 x 128 f32 output accumulator
#pragma unroll
  for (int nt = 0; nt < 8; ++nt) acc[nt] = v8f_zero();
  float mrow[8], lrow[8];
#pragma unroll
  for (int g = 0; g < 8; ++g) { mrow[g] = -3.0e38f; lrow[g] = 0.0f; }

  const float scale = rsqrtf((float)D);
  int kvLimit = CACHE + qrow0 + 16;            // exclusive bound on needed keys
  if (kvLimit > KV) kvLimit = KV;

#pragma clang loop unroll(disable)
  for (int kv0 = 0; kv0 < kvLimit; kv0 += 32) {
    // ---- scores: S = Q * K^T for 32 kv columns -------------------------
    v8f s0 = v8f_zero(), s1 = v8f_zero();
#pragma unroll
    for (int kk = 0; kk < 4; ++kk) {
      BFrag b0, b1;
      const unsigned short* kp0 = Kbase + (size_t)(kv0 + lr) * D + kk * 32 + lh * 8;
      const unsigned short* kp1 = Kbase + (size_t)(kv0 + 16 + lr) * D + kk * 32 + lh * 8;
      b0.q[0] = *(const v4u*)(kp0);  b0.q[1] = *(const v4u*)(kp0 + 16);
      b1.q[0] = *(const v4u*)(kp1);  b1.q[1] = *(const v4u*)(kp1 + 16);
      s0 = wmma_bf16(qa[kk].v, b0.v, s0);
      s1 = wmma_bf16(qa[kk].v, b1.v, s1);
    }

    // ---- scale + causal mask (key_pos <= CACHE + query_row) ------------
    const int qposb = CACHE + qrow0;
#pragma unroll
    for (int g = 0; g < 8; ++g) {
      int qpos = qposb + g + 8 * lh;
      s0[g] = (kv0 + lr      <= qpos) ? s0[g] * scale : -3.0e38f;
      s1[g] = (kv0 + 16 + lr <= qpos) ? s1[g] * scale : -3.0e38f;
    }

    // ---- online softmax ------------------------------------------------
#pragma unroll
    for (int g = 0; g < 8; ++g) {
      float bm = fmaxf(s0[g], s1[g]);
      bm = fmaxf(bm, __shfl_xor(bm, 1, 32));
      bm = fmaxf(bm, __shfl_xor(bm, 2, 32));
      bm = fmaxf(bm, __shfl_xor(bm, 4, 32));
      bm = fmaxf(bm, __shfl_xor(bm, 8, 32));   // row max within 16-lane half
      float mn = fmaxf(mrow[g], bm);
      float alpha = __expf(mrow[g] - mn);
      mrow[g] = mn;
      float p0 = __expf(s0[g] - mn);
      float p1 = __expf(s1[g] - mn);
      float rs = p0 + p1;
      rs += __shfl_xor(rs, 1, 32);
      rs += __shfl_xor(rs, 2, 32);
      rs += __shfl_xor(rs, 4, 32);
      rs += __shfl_xor(rs, 8, 32);
      lrow[g] = lrow[g] * alpha + rs;
      int row = g + 8 * lh;
      P[row * 32 + lr]      = f2bf(p0);        // stash P for layout change
      P[row * 32 + 16 + lr] = f2bf(p1);
#pragma unroll
      for (int nt = 0; nt < 8; ++nt) acc[nt][g] *= alpha;
    }

    // ---- reload P as an A fragment (DS ops are in-order per wave) ------
    BFrag pa;
    {
      const unsigned short* pp = P + (size_t)lr * 32;
      pa.q[0] = *(const v4u*)(pp + lh * 8);
      pa.q[1] = *(const v4u*)(pp + 16 + lh * 8);
    }

    // ---- PV: acc += P(16x32) * V(32x128) -------------------------------
#pragma unroll
    for (int nt = 0; nt < 8; ++nt) {
      BFrag vb;
      const unsigned short* vp = Vbase + (size_t)(nt * 16 + lr) * KV + kv0 + lh * 8;
      vb.q[0] = *(const v4u*)(vp);  vb.q[1] = *(const v4u*)(vp + 16);
      acc[nt] = wmma_bf16(pa.v, vb.v, acc[nt]);
    }
  }

  // ---- epilogue: normalize (8 reciprocals, 64 mults) and store bf16 ----
  float inv[8];
#pragma unroll
  for (int g = 0; g < 8; ++g) inv[g] = 1.0f / lrow[g];
#pragma unroll
  for (int nt = 0; nt < 8; ++nt)
#pragma unroll
    for (int g = 0; g < 8; ++g) {
      int row = g + 8 * lh;
      float o = acc[nt][g] * inv[g];
      size_t dst = ((size_t)bi * T + qrow0 + row) * HD + (size_t)hi * D + nt * 16 + lr;
      Y[dst] = f2bf(o);
    }
}

// ---------------------------------------------------------------------------
// Host launcher
// ---------------------------------------------------------------------------
extern "C" void kernel_launch(void* const* d_in, const int* in_sizes, int n_in,
                              void* d_out, int out_size, void* d_ws, size_t ws_size,
                              hipStream_t stream) {
  (void)in_sizes; (void)n_in; (void)out_size; (void)ws_size;

  const float* x       = (const float*)d_in[0];
  const float* cache_k = (const float*)d_in[1];
  const float* cache_v = (const float*)d_in[2];
  const float* Wq      = (const float*)d_in[3];
  const float* Wk      = (const float*)d_in[4];
  const float* Wv      = (const float*)d_in[5];
  const float* Wp      = (const float*)d_in[6];
  float* out = (float*)d_out;

  const int B = 2, T = 2048, C = 2048, H = 16, Dh = 128, CACHE = 2048, KV = 4096;
  const int M = B * T;

  // workspace carve-up (bf16 buffers), 256B aligned
  char* ws = (char*)d_ws;
  size_t off = 0;
  auto take = [&](size_t bytes) {
    char* p = ws + off;
    off += (bytes + 255) & ~(size_t)255;
    return p;
  };
  unsigned short* xb  = (unsigned short*)take((size_t)M * C * 2);     // x bf16
  unsigned short* wqb = (unsigned short*)take((size_t)C * C * 2);
  unsigned short* wkb = (unsigned short*)take((size_t)Dh * C * 2);
  unsigned short* wvb = (unsigned short*)take((size_t)Dh * C * 2);
  unsigned short* wpb = (unsigned short*)take((size_t)C * C * 2);
  unsigned short* qb  = (unsigned short*)take((size_t)M * C * 2);     // q bf16
  unsigned short* Kb  = (unsigned short*)take((size_t)B * KV * Dh * 2);
  unsigned short* Vtb = (unsigned short*)take((size_t)B * Dh * KV * 2);
  unsigned short* yb  = (unsigned short*)take((size_t)M * C * 2);     // attn out bf16

  // 1) converts
  {
    long long n;
    n = (long long)M * C;
    k_f32_to_bf16<<<(unsigned)((n + 255) / 256), 256, 0, stream>>>(x, xb, n);
    n = (long long)C * C;
    k_f32_to_bf16<<<(unsigned)((n + 255) / 256), 256, 0, stream>>>(Wq, wqb, n);
    k_f32_to_bf16<<<(unsigned)((n + 255) / 256), 256, 0, stream>>>(Wp, wpb, n);
    n = (long long)Dh * C;
    k_f32_to_bf16<<<(unsigned)((n + 255) / 256), 256, 0, stream>>>(Wk, wkb, n);
    k_f32_to_bf16<<<(unsigned)((n + 255) / 256), 256, 0, stream>>>(Wv, wvb, n);
    n = (long long)B * CACHE * Dh;
    k_cache_to_K <<<(unsigned)((n + 255) / 256), 256, 0, stream>>>(cache_k, Kb, CACHE, Dh, KV, n);
    k_cache_to_Vt<<<(unsigned)((n + 255) / 256), 256, 0, stream>>>(cache_v, Vtb, CACHE, Dh, KV, n);
  }

  // 2) projections (NT gemms against row-major weights)
  {
    dim3 blk(256);
    // q = x @ Wq^T  -> qb [b*t, 2048]
    dim3 gq(M / 64, (C + 255) / 256);
    gemm_bf16_wmma<false><<<gq, blk, 0, stream>>>(xb, C, wqb, C, qb, M, C, C,
        T, (long long)T * C, 0, C, 1);
    // k_new = x @ Wk^T -> Kb rows CACHE..CACHE+T-1 per batch
    dim3 gk(M / 64, 1);
    gemm_bf16_wmma<false><<<gk, blk, 0, stream>>>(xb, C, wkb, C, Kb, M, Dh, C,
        T, (long long)KV * Dh, (long long)CACHE * Dh, Dh, 1);
    // v_new = x @ Wv^T -> Vt[:, :, CACHE..] (transposed store via colStride)
    gemm_bf16_wmma<false><<<gk, blk, 0, stream>>>(xb, C, wvb, C, Vtb, M, Dh, C,
        T, (long long)Dh * KV, (long long)CACHE, 1, KV);
  }

  // 3) fused flash attention -> yb [b*t, 2048]
  {
    dim3 grid(T / 64, H, B);
    mqa_flash_attn<<<grid, 128, 0, stream>>>(qb, Kb, Vtb, yb, T, CACHE, KV, Dh, C);
  }

  // 4) output projection (f32 out) -> d_out
  {
    dim3 gp(M / 64, (C + 255) / 256);
    gemm_bf16_wmma<true><<<gp, dim3(256), 0, stream>>>(yb, C, wpb, C, out, M, C, C,
        T, (long long)T * C, 0, C, 1);
  }
}
